// HyperConv2d_31430570672477
// MI455X (gfx1250) — matrix operand order repaired and checked by reading, not compiled, not run on hardware
//
#include <hip/hip_runtime.h>
#include <stdint.h>

// Problem constants (match reference)
#define B_     32
#define INC    128
#define OUTC   128
#define ZD     64
#define H_     96
#define W_     96
#define HP     98      // padded height (1 halo each side)
#define WPAD   98      // padded width
#define KSTEPS 36      // 9 kernel taps * (128/32) channel slices

typedef __attribute__((ext_vector_type(16))) _Float16 v16h;
typedef __attribute__((ext_vector_type(8)))  float    v8f;
typedef int v4i_ __attribute__((vector_size(16)));   // matches builtin param type

// ---------------------------------------------------------------------------
// CDNA5 async global->LDS copy (ASYNCcnt path), with portable fallback.
// ---------------------------------------------------------------------------
#if defined(__AMDGCN__) && __has_builtin(__builtin_amdgcn_global_load_async_to_lds_b128)
#define USE_ASYNC 1
#else
#define USE_ASYNC 0
#endif

__device__ __forceinline__ void lds_copy16(void* ldsDst, const void* gSrc) {
#if USE_ASYNC
  __builtin_amdgcn_global_load_async_to_lds_b128(
      (__attribute__((address_space(1))) v4i_*)gSrc,
      (__attribute__((address_space(3))) v4i_*)ldsDst,
      /*offset=*/0, /*cpol=*/0);
#else
  *(uint4*)ldsDst = *(const uint4*)gSrc;
#endif
}

__device__ __forceinline__ void wait_async() {
#if USE_ASYNC
#if __has_builtin(__builtin_amdgcn_s_wait_asynccnt)
  __builtin_amdgcn_s_wait_asynccnt(0);
#else
  asm volatile("s_wait_asynccnt 0" ::: "memory");
#endif
#endif
}

// ---------------------------------------------------------------------------
// Kernel 1: x (f32 NCHW) -> padded NHWC f16  xp[b][hp][wp][c], hp/wp in [0,98)
// LDS-tiled 32x32 transpose so both global read and write are coalesced.
// Pad ring is pre-zeroed by hipMemsetAsync.
// ---------------------------------------------------------------------------
__global__ void xpack_kernel(const float* __restrict__ x, _Float16* __restrict__ xp) {
  __shared__ float tile[32][33];
  const int b  = blockIdx.z;
  const int h  = blockIdx.y;            // unpadded row 0..95
  const int wt = blockIdx.x >> 2;       // 0..2  (w tile of 32)
  const int ct = blockIdx.x & 3;        // 0..3  (c tile of 32)
  const int w0 = wt * 32, c0 = ct * 32;
  const int t  = threadIdx.x;
  const int tw = t & 31, tr = t >> 5;   // 8 rows of 32 lanes

#pragma unroll
  for (int k = 0; k < 4; ++k) {
    const int c = tr + 8 * k;
    tile[c][tw] = x[(((size_t)b * INC + (c0 + c)) * H_ + h) * W_ + (w0 + tw)];
  }
  __syncthreads();

  const int cc = t & 31, wr = t >> 5;
#pragma unroll
  for (int k = 0; k < 4; ++k) {
    const int w = wr + 8 * k;
    xp[(((size_t)b * HP + (h + 1)) * WPAD + (w0 + w + 1)) * INC + (c0 + cc)] =
        (_Float16)tile[cc][w];
  }
}

// ---------------------------------------------------------------------------
// Kernel 2: merged hyper-weights -> f16, pre-swizzled for the GEMM A operand:
//   wsA[b][s][m][kk], s = (kh*3+kw)*4 + (i>>5), m = o, kk = i&31
// One block per (b, o), 128 threads (thread = input channel i).
// ---------------------------------------------------------------------------
__global__ void wgen_kernel(const float* __restrict__ z,
                            const float* __restrict__ basew,   // [O][I][3][3]
                            const float* __restrict__ gatew,   // [O][64]
                            const float* __restrict__ pww,     // [O*I][64]
                            const float* __restrict__ dww,     // [O*9][64]
                            _Float16* __restrict__ wsA) {
  const int b = blockIdx.x >> 7;
  const int o = blockIdx.x & 127;
  __shared__ float zs[ZD];
  __shared__ float gdw[10];              // [0]=gate, [1..9]=dw taps
  const int t = threadIdx.x;

  if (t < ZD) zs[t] = z[(size_t)b * ZD + t];
  __syncthreads();

  if (t < 10) {
    const float* wr = (t == 0) ? (gatew + (size_t)o * ZD)
                               : (dww + ((size_t)o * 9 + (t - 1)) * ZD);
    float s = 0.f;
    for (int j = 0; j < ZD; ++j) s += zs[j] * wr[j];
    gdw[t] = s;
  }
  __syncthreads();

  const float* pr = pww + ((size_t)o * INC + t) * ZD;
  float pw = 0.f;
  for (int j = 0; j < ZD; ++j) pw += zs[j] * pr[j];
  const float gate = gdw[0];

#pragma unroll
  for (int kp = 0; kp < 9; ++kp) {
    const float bw = basew[((size_t)o * INC + t) * 9 + kp];
    const float wv = bw * gate + pw * gdw[1 + kp];
    const int s = kp * 4 + (t >> 5);
    wsA[(((size_t)b * KSTEPS + s) * OUTC + o) * 32 + (t & 31)] = (_Float16)wv;
  }
}

// ---------------------------------------------------------------------------
// Kernel 3: implicit-GEMM conv. One block per (b, row h): M=128, N=96, K=1152.
// 8 waves: 4 along M (32 each) x 2 along N (48 each) -> 2x3 accum frags/wave.
// Double-buffered LDS staged with async-to-LDS B128; v_wmma_f32_16x16x32_f16.
// ---------------------------------------------------------------------------
union FragH { v16h v; unsigned int u[8]; };

// A fragment: 16x32 f16, rows = output channels. LDS layout: [m][32k] rows.
__device__ __forceinline__ v16h load_fragA(const _Float16* A, int m0, int lane) {
  FragH f;
  const unsigned int* row = (const unsigned int*)(A + (size_t)(m0 + (lane & 15)) * 32);
  const int half8 = (lane >> 4) << 3;  // lanes 16-31 -> K base +8
#pragma unroll
  for (int v = 0; v < 8; ++v) {
    const int kb = ((v & 4) << 2) + half8 + ((v & 3) << 1);  // even K index
    f.u[v] = row[kb >> 1];
  }
  return f.v;
}

// B fragment: 32x16 f16, columns = pixels. LDS layout: [n][32k] rows.
__device__ __forceinline__ v16h load_fragB(const _Float16* Bt, int n0, int lane) {
  FragH f;
  const unsigned int* row = (const unsigned int*)(Bt + (size_t)(n0 + (lane & 15)) * 32);
  const int dbase = (lane >> 4) << 3;  // lanes 16-31 -> K base +8
#pragma unroll
  for (int v = 0; v < 8; ++v) f.u[v] = row[((v & 4) << 1) + (dbase >> 1) + (v & 3)];
  return f.v;
}

__global__ void __launch_bounds__(256, 1)
hyperconv_kernel(const _Float16* __restrict__ xp,   // [B][98][98][128] f16
                 const _Float16* __restrict__ wA,   // [B][36][128][32] f16
                 float* __restrict__ out) {         // [B][128][96][96] f32
  __shared__ alignas(16) _Float16 Abuf[2][OUTC * 32];  // 2 x 8 KB
  __shared__ alignas(16) _Float16 Bbuf[2][W_ * 32];    // 2 x 6 KB

  const int h = blockIdx.x;
  const int b = blockIdx.y;
  const int t = threadIdx.x;
  const int lane = t & 31;
  const int wave = t >> 5;
  const int m0 = (wave & 3) * 32;   // wave's M offset (2 frags of 16)
  const int n0 = (wave >> 2) * 48;  // wave's N offset (3 frags of 16)

  const _Float16* Abase = wA + (size_t)b * KSTEPS * OUTC * 32;

  v8f acc[2][3];
  const v8f vzero = {0.f, 0.f, 0.f, 0.f, 0.f, 0.f, 0.f, 0.f};
#pragma unroll
  for (int i = 0; i < 2; ++i)
#pragma unroll
    for (int j = 0; j < 3; ++j) acc[i][j] = vzero;

  auto stage = [&](int s, int nb) {
    // A tile: 8192 contiguous bytes -> 512 B128 chunks, 2 per thread.
    const char* srcA = (const char*)(Abase + (size_t)s * OUTC * 32);
    char* dstA = (char*)Abuf[nb];
    lds_copy16(dstA + t * 16, srcA + t * 16);
    lds_copy16(dstA + 4096 + t * 16, srcA + 4096 + t * 16);
    // B tile: 96 rows x 64 B (32 channels f16) -> 384 B128 chunks.
    const int kp = s >> 2, i0 = (s & 3) << 5;
    const int kh = kp / 3, kw = kp % 3;
    const _Float16* rowbase = xp + ((size_t)b * HP + (h + kh)) * WPAD * INC;
    {
      const int n = t >> 2, part = t & 3;
      lds_copy16((char*)Bbuf[nb] + n * 64 + part * 16,
                 (const char*)(rowbase + (size_t)(n + kw) * INC + i0 + part * 8));
    }
    if (t < 128) {
      const int c = t + 256;
      const int n = c >> 2, part = c & 3;
      lds_copy16((char*)Bbuf[nb] + n * 64 + part * 16,
                 (const char*)(rowbase + (size_t)(n + kw) * INC + i0 + part * 8));
    }
  };

  int cur = 0;
  stage(0, 0);
  wait_async();
  __syncthreads();

  for (int s = 0; s < KSTEPS; ++s) {
    if (s + 1 < KSTEPS) stage(s + 1, cur ^ 1);  // overlap with compute below

    const _Float16* A  = Abuf[cur];
    const _Float16* Bt = Bbuf[cur];
    const v16h a0 = load_fragA(A, m0, lane);
    const v16h a1 = load_fragA(A, m0 + 16, lane);
    const v16h b0 = load_fragB(Bt, n0, lane);
    const v16h b1 = load_fragB(Bt, n0 + 16, lane);
    const v16h b2 = load_fragB(Bt, n0 + 32, lane);

    acc[0][0] = __builtin_amdgcn_wmma_f32_16x16x32_f16(false, a0, false, b0, (short)0, acc[0][0], false, false);
    acc[0][1] = __builtin_amdgcn_wmma_f32_16x16x32_f16(false, a0, false, b1, (short)0, acc[0][1], false, false);
    acc[0][2] = __builtin_amdgcn_wmma_f32_16x16x32_f16(false, a0, false, b2, (short)0, acc[0][2], false, false);
    acc[1][0] = __builtin_amdgcn_wmma_f32_16x16x32_f16(false, a1, false, b0, (short)0, acc[1][0], false, false);
    acc[1][1] = __builtin_amdgcn_wmma_f32_16x16x32_f16(false, a1, false, b1, (short)0, acc[1][1], false, false);
    acc[1][2] = __builtin_amdgcn_wmma_f32_16x16x32_f16(false, a1, false, b2, (short)0, acc[1][2], false, false);

    wait_async();
    __syncthreads();
    cur ^= 1;
  }

  // D layout: VGPR r -> M = r + (lane>=16 ? 8 : 0), N = lane & 15.
  const int nlo = lane & 15;
  const int hi8 = (lane >> 4) << 3;
#pragma unroll
  for (int mi = 0; mi < 2; ++mi)
#pragma unroll
    for (int ni = 0; ni < 3; ++ni)
#pragma unroll
      for (int r = 0; r < 8; ++r) {
        const int o = m0 + mi * 16 + hi8 + r;
        const int w = n0 + ni * 16 + nlo;
        out[(((size_t)b * OUTC + o) * H_ + h) * W_ + w] = acc[mi][ni][r];
      }
}

// ---------------------------------------------------------------------------
// Host launcher
// ---------------------------------------------------------------------------
extern "C" void kernel_launch(void* const* d_in, const int* in_sizes, int n_in,
                              void* d_out, int out_size, void* d_ws, size_t ws_size,
                              hipStream_t stream) {
  (void)in_sizes; (void)n_in; (void)out_size; (void)ws_size;
  const float* x     = (const float*)d_in[0];
  const float* z     = (const float*)d_in[1];
  const float* basew = (const float*)d_in[2];
  const float* gatew = (const float*)d_in[3];
  const float* pww   = (const float*)d_in[4];
  const float* dww   = (const float*)d_in[5];
  float* out = (float*)d_out;

  const size_t xpBytes = (size_t)B_ * HP * WPAD * INC * 2;  // f16 padded NHWC
  _Float16* xp  = (_Float16*)d_ws;
  _Float16* wsA = (_Float16*)((char*)d_ws + xpBytes);

  // Zero the halo ring (interior is fully overwritten each call).
  (void)hipMemsetAsync(d_ws, 0, xpBytes, stream);

  xpack_kernel<<<dim3(12, 96, B_), 256, 0, stream>>>(x, xp);
  wgen_kernel<<<dim3(B_ * OUTC), 128, 0, stream>>>(z, basew, gatew, pww, dww, wsA);
  hyperconv_kernel<<<dim3(H_, B_), 256, 0, stream>>>(xp, wsA, out);
}
